// MultiheadAttention_76991583748350
// MI455X (gfx1250) — matrix operand order repaired and checked by reading, not compiled, not run on hardware
//
#include <hip/hip_runtime.h>
#include <hip/hip_bf16.h>

// ---------------------------------------------------------------------------
// MHA forward for MI455X (gfx1250, wave32, WMMA).
//   B=4, S=2048, N_DIM=1024, HEADS=16, HEAD_DIM=64
// Pipeline:
//   k1: Q = x1*Wq, K = x2*Wk, V = x2*Wv   (bf16 out; V stored transposed [b,h,d,s])
//   k2: flash attention per (b, h, 64 q-rows), online softmax, causal+pad masks
//   k3: out = ctx*Wo + bo  (fp32 out)
// GEMMs: 128x128 block tile, 8 waves x (32x64) each, double-buffered LDS with
// software-pipelined global->register staging (loads overlap WMMA).
// All matrix math: v_wmma_f32_16x16x32_bf16 (fp32 accumulate).
// ---------------------------------------------------------------------------

#define NDIM   1024
#define HEADS  16
#define HD     64
#define SEQ    2048
#define BATCH  4

typedef __bf16 bf16_t;
typedef __attribute__((ext_vector_type(16))) __bf16 v16bf;
typedef __attribute__((ext_vector_type(8)))  __bf16 v8bf;
typedef __attribute__((ext_vector_type(8)))  float  v8f;

static __device__ __forceinline__ v16bf join16(v8bf lo, v8bf hi) {
  return __builtin_shufflevector(lo, hi, 0,1,2,3,4,5,6,7,8,9,10,11,12,13,14,15);
}
static __device__ __forceinline__ v8bf ld8(const bf16_t* p) {
  return *(const v8bf*)p;
}
static __device__ __forceinline__ v8bf cvt8(float4 a, float4 b) {
  v8bf r;
  r[0]=(bf16_t)a.x; r[1]=(bf16_t)a.y; r[2]=(bf16_t)a.z; r[3]=(bf16_t)a.w;
  r[4]=(bf16_t)b.x; r[5]=(bf16_t)b.y; r[6]=(bf16_t)b.z; r[7]=(bf16_t)b.w;
  return r;
}
static __device__ __forceinline__ v8f wmma_bf16(v16bf a, v16bf b, v8f c) {
  return __builtin_amdgcn_wmma_f32_16x16x32_bf16(false, a, false, b, (short)0, c,
                                                 false, false);
}
static __device__ __forceinline__ float rowmax16(float x) {
  x = fmaxf(x, __shfl_xor(x, 1, 32));
  x = fmaxf(x, __shfl_xor(x, 2, 32));
  x = fmaxf(x, __shfl_xor(x, 4, 32));
  x = fmaxf(x, __shfl_xor(x, 8, 32));
  return x;
}
static __device__ __forceinline__ float rowsum16(float x) {
  x += __shfl_xor(x, 1, 32);
  x += __shfl_xor(x, 2, 32);
  x += __shfl_xor(x, 4, 32);
  x += __shfl_xor(x, 8, 32);
  return x;
}

// ---------------------------------------------------------------------------
// Kernel 1: fused QKV projection.
// grid = (SEQ/128, NDIM/128, 3*BATCH), block = 256 (8 waves, 4x2 wave grid).
// Block: 128 rows x 128 cols (2 heads) of one of Q/K/V for one batch.
// Double-buffered LDS; next K-tile global loads issued before the barrier so
// they overlap the 8 WMMAs of the current tile.
// ---------------------------------------------------------------------------
__global__ __launch_bounds__(256) void qkv_proj_kernel(
    const float* __restrict__ x1, const float* __restrict__ x2,
    const float* __restrict__ Wq, const float* __restrict__ Wk,
    const float* __restrict__ Wv,
    bf16_t* __restrict__ Qb, bf16_t* __restrict__ Kb, bf16_t* __restrict__ Vt)
{
  __shared__ bf16_t As[2][128][40];  // [m][k], k-contiguous, +8 pad (16B rows)
  __shared__ bf16_t Bt[2][128][40];  // [n][k], k-contiguous (W column tile)

  const int mat  = blockIdx.z % 3;       // 0:Q 1:K 2:V
  const int b    = blockIdx.z / 3;
  const int row0 = blockIdx.x * 128;     // row within batch
  const int n0g  = blockIdx.y * 128;     // global output column base (2 heads)

  const float* xin = (mat == 0 ? x1 : x2) + (size_t)b * SEQ * NDIM;
  const float* W   = (mat == 0 ? Wq : (mat == 1 ? Wk : Wv));

  const int tid = threadIdx.x;
  const int wave = tid >> 5, lane = tid & 31;
  const int lm = lane & 15, lh = lane >> 4;
  const int m0 = (wave >> 1) * 32;       // 0,32,64,96
  const int n0 = (wave & 1) * 64;        // 0,64

  // staging thread mapping: A 128x32 (16 f32/thr), B 128x32 (16 f32/thr strided)
  const int ar_ = tid >> 1, akh = (tid & 1) * 16;
  const int bn_ = tid & 127, bkb = (tid >> 7) * 16;
  const float* aptr = xin + (size_t)(row0 + ar_) * NDIM + akh;
  const float* bptr = W + (size_t)bkb * NDIM + n0g + bn_;

  float4 a0, a1, a2, a3;
  float  br[16];
  {   // prologue: load K-tile 0 into registers
    const float4* s4 = (const float4*)aptr;
    a0 = s4[0]; a1 = s4[1]; a2 = s4[2]; a3 = s4[3];
    #pragma unroll
    for (int i = 0; i < 16; ++i) br[i] = bptr[(size_t)i * NDIM];
  }

  v8f acc[2][4] = {};
  const int NIT = NDIM / 32;
  for (int it = 0; it < NIT; ++it) {
    const int buf = it & 1;
    // convert staged registers -> LDS tile `buf`
    *(v8bf*)&As[buf][ar_][akh]     = cvt8(a0, a1);
    *(v8bf*)&As[buf][ar_][akh + 8] = cvt8(a2, a3);
    {
      v8bf p0, p1;
      #pragma unroll
      for (int i = 0; i < 8; ++i) { p0[i] = (bf16_t)br[i]; p1[i] = (bf16_t)br[i + 8]; }
      *(v8bf*)&Bt[buf][bn_][bkb]     = p0;
      *(v8bf*)&Bt[buf][bn_][bkb + 8] = p1;
    }
    // issue next tile's global loads (wait lands next iteration, after WMMAs)
    if (it + 1 < NIT) {
      const float4* s4 = (const float4*)(aptr + (it + 1) * 32);
      a0 = s4[0]; a1 = s4[1]; a2 = s4[2]; a3 = s4[3];
      const float* wp = bptr + (size_t)(it + 1) * 32 * NDIM;
      #pragma unroll
      for (int i = 0; i < 16; ++i) br[i] = wp[(size_t)i * NDIM];
      if (it + 2 < NIT) __builtin_prefetch(aptr + (it + 2) * 32, 0, 3);
    }
    __syncthreads();

    // ---- 2x4 tiles of 16x16: 8 WMMAs per wave per K-step ----
    v16bf af[2], bfr[4];
    #pragma unroll
    for (int mt = 0; mt < 2; ++mt) {
      const bf16_t* ap = &As[buf][m0 + mt * 16 + lm][0];
      af[mt] = join16(ld8(ap + lh * 8), ld8(ap + 16 + lh * 8));
    }
    #pragma unroll
    for (int nt = 0; nt < 4; ++nt) {
      const bf16_t* bp = &Bt[buf][n0 + nt * 16 + lm][0];
      bfr[nt] = join16(ld8(bp + lh * 8), ld8(bp + 16 + lh * 8));
    }
    #pragma unroll
    for (int mt = 0; mt < 2; ++mt)
      #pragma unroll
      for (int nt = 0; nt < 4; ++nt)
        acc[mt][nt] = wmma_bf16(af[mt], bfr[nt], acc[mt][nt]);
  }

  // ---- store ----
  #pragma unroll
  for (int mt = 0; mt < 2; ++mt) {
    #pragma unroll
    for (int nt = 0; nt < 4; ++nt) {
      const int col  = n0g + n0 + nt * 16 + lm;   // global out column
      const int head = col >> 6, d = col & 63;
      const size_t bh = (size_t)(b * HEADS + head);
      if (mat == 2) {
        // V transposed [b,h,d,s]: lane holds fixed d, 8 consecutive s -> b128
        const int s0 = row0 + m0 + mt * 16 + lh * 8;
        v8bf pk;
        #pragma unroll
        for (int v = 0; v < 8; ++v) pk[v] = (bf16_t)acc[mt][nt][v];
        *(v8bf*)(Vt + (bh * HD + d) * SEQ + s0) = pk;
      } else {
        bf16_t* outp = (mat == 0 ? Qb : Kb);
        #pragma unroll
        for (int v = 0; v < 8; ++v) {
          const int srow = row0 + m0 + mt * 16 + lh * 8 + v;
          outp[(bh * SEQ + srow) * HD + d] = (bf16_t)acc[mt][nt][v];
        }
      }
    }
  }
}

// ---------------------------------------------------------------------------
// Kernel 2: flash attention. grid = (SEQ/64, HEADS, BATCH), block = 128.
// Wave w owns q-rows [qrow0, qrow0+16); loops kv in chunks of 32 up to its
// own causal extent (no cross-wave sync inside the divergent loop).
// ---------------------------------------------------------------------------
__global__ __launch_bounds__(128) void attn_kernel(
    const bf16_t* __restrict__ Qb, const bf16_t* __restrict__ Kb,
    const bf16_t* __restrict__ Vt, const unsigned char* __restrict__ x2_mask,
    bf16_t* __restrict__ ctx)
{
  __shared__ bf16_t Pbuf[4][16][40];   // per-wave P transpose scratch (Dlyt->Alyt)

  const int b = blockIdx.z, h = blockIdx.y;
  const int tid = threadIdx.x;
  const int wave = tid >> 5, lane = tid & 31;
  const int lm = lane & 15, lh = lane >> 4;
  const int qrow0 = blockIdx.x * 64 + wave * 16;

  const size_t qkBase = (size_t)(b * HEADS + h) * SEQ * HD;   // [b,h,s,d]
  const size_t vBase  = (size_t)(b * HEADS + h) * HD * SEQ;   // [b,h,d,s]

  // Q fragments (held in registers for the whole kv loop)
  const bf16_t* qp = Qb + qkBase + (size_t)(qrow0 + lm) * HD;
  const v16bf qf0 = join16(ld8(qp + lh * 8),      ld8(qp + 16 + lh * 8));
  const v16bf qf1 = join16(ld8(qp + 32 + lh * 8), ld8(qp + 48 + lh * 8));

  v8f   O[4] = {};
  float mi[8], li[8];
  #pragma unroll
  for (int v = 0; v < 8; ++v) { mi[v] = -__builtin_inff(); li[v] = 0.f; }

  const int jend = qrow0 + 16;          // exclusive causal limit (<= SEQ)
  for (int j = 0; j < jend; j += 32) {  // j <= SEQ-32, so all loads in-bounds
    // ---- S = Q * K^T for kv cols [j, j+32) ----
    v8f st[2] = {};
    #pragma unroll
    for (int t = 0; t < 2; ++t) {
      const bf16_t* kp = Kb + qkBase + (size_t)(j + t * 16 + lm) * HD;
      const v16bf kf0 = join16(ld8(kp + lh * 8),      ld8(kp + 16 + lh * 8));
      const v16bf kf1 = join16(ld8(kp + 32 + lh * 8), ld8(kp + 48 + lh * 8));
      st[t] = wmma_bf16(qf0, kf0, st[t]);
      st[t] = wmma_bf16(qf1, kf1, st[t]);
    }

    // ---- scale, multiplicative pad mask, causal -inf (triu(k=1)) ----
    #pragma unroll
    for (int t = 0; t < 2; ++t) {
      const int col = j + t * 16 + lm;
      const float keep = x2_mask[b * SEQ + col] ? 0.f : 1.f;
      #pragma unroll
      for (int v = 0; v < 8; ++v) {
        const int row = qrow0 + lh * 8 + v;
        float s = st[t][v] * 0.125f * keep;        // 1/sqrt(64)
        if (col > row) s = -__builtin_inff();
        st[t][v] = s;
      }
    }

    // ---- online softmax ----
    float alpha[8], rsum[8];
    #pragma unroll
    for (int v = 0; v < 8; ++v) {
      const float rm   = rowmax16(fmaxf(st[0][v], st[1][v]));
      const float mnew = fmaxf(mi[v], rm);
      alpha[v] = __expf(mi[v] - mnew);
      mi[v] = mnew;
      rsum[v] = 0.f;
    }
    #pragma unroll
    for (int t = 0; t < 2; ++t)
      #pragma unroll
      for (int v = 0; v < 8; ++v) {
        const float p = __expf(st[t][v] - mi[v]);
        st[t][v] = p;
        rsum[v] += p;
      }
    #pragma unroll
    for (int v = 0; v < 8; ++v) {
      li[v] = li[v] * alpha[v] + rowsum16(rsum[v]);
      #pragma unroll
      for (int dt = 0; dt < 4; ++dt) O[dt][v] *= alpha[v];
    }

    // ---- transpose P (D-layout) -> A-layout via per-wave LDS ----
    #pragma unroll
    for (int t = 0; t < 2; ++t)
      #pragma unroll
      for (int v = 0; v < 8; ++v)
        Pbuf[wave][lh * 8 + v][t * 16 + lm] = (bf16_t)st[t][v];
    asm volatile("s_wait_dscnt 0x0" ::: "memory");
    __builtin_amdgcn_wave_barrier();
    const bf16_t* pr = &Pbuf[wave][lm][0];
    const v16bf pf = join16(ld8(pr + lh * 16), ld8(pr + lh * 16 + 8));
    __builtin_amdgcn_wave_barrier();

    // ---- O += P * V  (V transposed: contiguous 16 kv per lane) ----
    #pragma unroll
    for (int dt = 0; dt < 4; ++dt) {
      const bf16_t* vp = Vt + vBase + (size_t)(dt * 16 + lm) * SEQ + j + lh * 16;
      const v16bf vf = join16(ld8(vp), ld8(vp + 8));
      O[dt] = wmma_bf16(pf, vf, O[dt]);
    }
  }

  // ---- normalize and store ctx[b][s][h*64+d] (bf16) ----
  float inv[8];
  #pragma unroll
  for (int v = 0; v < 8; ++v) inv[v] = 1.f / li[v];
  #pragma unroll
  for (int dt = 0; dt < 4; ++dt) {
    const int d = dt * 16 + lm;
    #pragma unroll
    for (int v = 0; v < 8; ++v) {
      const int srow = qrow0 + lh * 8 + v;
      ctx[((size_t)b * SEQ + srow) * NDIM + h * HD + d] = (bf16_t)(O[dt][v] * inv[v]);
    }
  }
}

// ---------------------------------------------------------------------------
// Kernel 3: out = ctx * Wo + bo (fp32).
// grid = (B*SEQ/128, NDIM/128), block = 256. Same pipelined GEMM as kernel 1.
// ---------------------------------------------------------------------------
__global__ __launch_bounds__(256) void out_proj_kernel(
    const bf16_t* __restrict__ ctx, const float* __restrict__ Wo,
    const float* __restrict__ bo, float* __restrict__ out)
{
  __shared__ bf16_t As[2][128][40];
  __shared__ bf16_t Bt[2][128][40];

  const int row0 = blockIdx.x * 128;
  const int n0g  = blockIdx.y * 128;
  const int tid  = threadIdx.x;
  const int wave = tid >> 5, lane = tid & 31;
  const int lm = lane & 15, lh = lane >> 4;
  const int m0 = (wave >> 1) * 32, n0 = (wave & 1) * 64;

  const int ar_ = tid >> 1, akh = (tid & 1) * 16;
  const int bn_ = tid & 127, bkb = (tid >> 7) * 16;
  const bf16_t* aptr = ctx + (size_t)(row0 + ar_) * NDIM + akh;
  const float*  bptr = Wo + (size_t)bkb * NDIM + n0g + bn_;

  v8bf  alo, ahi;
  float br[16];
  {   // prologue
    alo = ld8(aptr); ahi = ld8(aptr + 8);
    #pragma unroll
    for (int i = 0; i < 16; ++i) br[i] = bptr[(size_t)i * NDIM];
  }

  v8f acc[2][4] = {};
  const int NIT = NDIM / 32;
  for (int it = 0; it < NIT; ++it) {
    const int buf = it & 1;
    *(v8bf*)&As[buf][ar_][akh]     = alo;
    *(v8bf*)&As[buf][ar_][akh + 8] = ahi;
    {
      v8bf p0, p1;
      #pragma unroll
      for (int i = 0; i < 8; ++i) { p0[i] = (bf16_t)br[i]; p1[i] = (bf16_t)br[i + 8]; }
      *(v8bf*)&Bt[buf][bn_][bkb]     = p0;
      *(v8bf*)&Bt[buf][bn_][bkb + 8] = p1;
    }
    if (it + 1 < NIT) {
      alo = ld8(aptr + (it + 1) * 32);
      ahi = ld8(aptr + (it + 1) * 32 + 8);
      const float* wp = bptr + (size_t)(it + 1) * 32 * NDIM;
      #pragma unroll
      for (int i = 0; i < 16; ++i) br[i] = wp[(size_t)i * NDIM];
      if (it + 2 < NIT) __builtin_prefetch(aptr + (it + 2) * 32, 0, 3);
    }
    __syncthreads();

    v16bf af[2], bfr[4];
    #pragma unroll
    for (int mt = 0; mt < 2; ++mt) {
      const bf16_t* ap = &As[buf][m0 + mt * 16 + lm][0];
      af[mt] = join16(ld8(ap + lh * 8), ld8(ap + 16 + lh * 8));
    }
    #pragma unroll
    for (int nt = 0; nt < 4; ++nt) {
      const bf16_t* bp = &Bt[buf][n0 + nt * 16 + lm][0];
      bfr[nt] = join16(ld8(bp + lh * 8), ld8(bp + 16 + lh * 8));
    }
    #pragma unroll
    for (int mt = 0; mt < 2; ++mt)
      #pragma unroll
      for (int nt = 0; nt < 4; ++nt)
        acc[mt][nt] = wmma_bf16(af[mt], bfr[nt], acc[mt][nt]);
  }

  #pragma unroll
  for (int mt = 0; mt < 2; ++mt)
    #pragma unroll
    for (int nt = 0; nt < 4; ++nt) {
      const int col = n0g + n0 + nt * 16 + lm;
      const float bias = bo[col];
      #pragma unroll
      for (int v = 0; v < 8; ++v) {
        const int r = row0 + m0 + mt * 16 + lh * 8 + v;
        out[(size_t)r * NDIM + col] = acc[mt][nt][v] + bias;
      }
    }
}

// ---------------------------------------------------------------------------
// Host launcher.
// inputs: 0:x1 1:x2 2:x2_mask(bool) 3:attn_mask(unused, computed analytically)
//         4:Wq 5:Wk 6:Wv 7:Wo 8:bo
// workspace (bf16): Q | K | V^T | ctx  = 4 * 8,388,608 elems = 64 MB
// ---------------------------------------------------------------------------
extern "C" void kernel_launch(void* const* d_in, const int* in_sizes, int n_in,
                              void* d_out, int out_size, void* d_ws, size_t ws_size,
                              hipStream_t stream) {
  (void)in_sizes; (void)n_in; (void)out_size; (void)ws_size;
  const float* x1 = (const float*)d_in[0];
  const float* x2 = (const float*)d_in[1];
  const unsigned char* x2_mask = (const unsigned char*)d_in[2];  // numpy bool = 1 byte
  const float* Wq = (const float*)d_in[4];
  const float* Wk = (const float*)d_in[5];
  const float* Wv = (const float*)d_in[6];
  const float* Wo = (const float*)d_in[7];
  const float* bo = (const float*)d_in[8];

  const size_t QKV = (size_t)BATCH * HEADS * SEQ * HD;  // 8,388,608
  bf16_t* Qb  = (bf16_t*)d_ws;
  bf16_t* Kb  = Qb + QKV;
  bf16_t* Vt  = Kb + QKV;
  bf16_t* ctx = Vt + QKV;

  dim3 gA(SEQ / 128, NDIM / 128, 3 * BATCH);
  qkv_proj_kernel<<<gA, 256, 0, stream>>>(x1, x2, Wq, Wk, Wv, Qb, Kb, Vt);

  dim3 gB(SEQ / 64, HEADS, BATCH);
  attn_kernel<<<gB, 128, 0, stream>>>(Qb, Kb, Vt, x2_mask, ctx);

  dim3 gC((BATCH * SEQ) / 128, NDIM / 128, 1);
  out_proj_kernel<<<gC, 256, 0, stream>>>(ctx, Wo, bo, (float*)d_out);
}
